// HeteroGAT_11235634446918
// MI455X (gfx1250) — matrix operand order repaired and checked by reading, not compiled, not run on hardware
//
#include <hip/hip_runtime.h>
#include <math.h>

#define N_REQ 20000
#define N_VEH 2000
#define E_RR 200000
#define E_VR 100000
#define E_RV 100000
#define NEG_SLOPE 0.2f

typedef __attribute__((ext_vector_type(16))) _Float16 v16h;
typedef __attribute__((ext_vector_type(8)))  float    v8f;

// ---------------------------------------------------------------------------
// WMMA GEMM: C[M,256] (f16) = A[M,K] (f32) @ W[K,256] (f32), K in {32,64,256}
// Block: 256 threads (8 waves). Tile: 32 rows x 256 cols, K-step 32.
// Wave (wid): rows (wid&1)*16..+16, cols (wid>>1)*64..+64 (4 WMMA n-tiles).
// A staged to LDS with contiguous k so each lane's v16h fragment
// ({K 0..15} or {K 16..31}) is one contiguous 32B ds load.
// W staged transposed (Wlds[n][k]) so B fragments are contiguous too.
// ---------------------------------------------------------------------------
__global__ void wmma_gemm_f16(const float* __restrict__ A,
                              const float* __restrict__ W,
                              _Float16* __restrict__ C, int M, int K) {
  __shared__ __align__(32) _Float16 Alds[32 * 32];
  __shared__ __align__(32) _Float16 Wlds[256 * 32];
  const int t     = threadIdx.x;
  const int lane  = t & 31;
  const int wid   = t >> 5;
  const int m_w   = (wid & 1) * 16;
  const int n_wb  = (wid >> 1) * 64;
  const int row0  = blockIdx.x * 32;
  const int hsel  = (lane < 16) ? 0 : 16;   // fragment half-select (in halfs)
  const int mA    = m_w + (lane & 15);
  const int ksteps = K >> 5;

  v8f acc[4];
#pragma unroll
  for (int i = 0; i < 4; ++i)
#pragma unroll
    for (int r = 0; r < 8; ++r) acc[i][r] = 0.0f;

  for (int ks = 0; ks < ksteps; ++ks) {
    const int k0 = ks << 5;
    {  // stage A: 32x32, fp32->f16, contiguous k
      const int m    = t >> 3;
      const int kq   = (t & 7) << 2;
      const int slot = kq;
      float4 v = make_float4(0.f, 0.f, 0.f, 0.f);
      const int row = row0 + m;
      if (row < M)
        v = *reinterpret_cast<const float4*>(A + (size_t)row * K + k0 + kq);
      _Float16* dp = &Alds[m * 32 + slot];
      dp[0] = (_Float16)v.x; dp[1] = (_Float16)v.y;
      dp[2] = (_Float16)v.z; dp[3] = (_Float16)v.w;
    }
    {  // stage W transposed: Wlds[n*32+k], n = t
#pragma unroll
      for (int k = 0; k < 32; ++k)
        Wlds[t * 32 + k] = (_Float16)W[(size_t)(k0 + k) * 256 + t];
    }
    __syncthreads();

    v16h a = *reinterpret_cast<const v16h*>(&Alds[mA * 32 + hsel]);
#pragma unroll
    for (int nt = 0; nt < 4; ++nt) {
      const int n = n_wb + nt * 16 + (lane & 15);
      v16h b = *reinterpret_cast<const v16h*>(&Wlds[n * 32 + hsel]);
      acc[nt] = __builtin_amdgcn_wmma_f32_16x16x32_f16(
          false, a, false, b, (short)0, acc[nt], false, false);
    }
    __syncthreads();
  }

  // C/D layout: VGPR r -> lanes 0-15: M=r, lanes 16-31: M=8+r; N = lane&15
#pragma unroll
  for (int nt = 0; nt < 4; ++nt) {
    const int col = n_wb + nt * 16 + (lane & 15);
#pragma unroll
    for (int r = 0; r < 8; ++r) {
      const int row = row0 + m_w + r + ((lane < 16) ? 0 : 8);
      if (row < M) C[(size_t)row * 256 + col] = (_Float16)acc[nt][r];
    }
  }
}

// ---------------------------------------------------------------------------
// Fold attention vector into weight: outf[k,h] = sum_c w[k, h*128+c]*att[h,c]
// ---------------------------------------------------------------------------
__global__ void fold_att(const float* __restrict__ w, const float* __restrict__ att,
                         float* __restrict__ outf, int K) {
  int k = blockIdx.x * blockDim.x + threadIdx.x;
  if (k >= K) return;
  const float* wr = w + (size_t)k * 256;
  float s0 = 0.f, s1 = 0.f;
  for (int c = 0; c < 128; ++c) {
    s0 += wr[c] * att[c];
    s1 += wr[128 + c] * att[128 + c];
  }
  outf[k * 2] = s0; outf[k * 2 + 1] = s1;
}

// alpha[n,h] = x[n,:] @ wf[:,h]   (wf is the folded [K,2] matrix)
__global__ void node_alpha(const float* __restrict__ x, const float* __restrict__ wf,
                           float* __restrict__ alpha, int N, int K) {
  int n = blockIdx.x * blockDim.x + threadIdx.x;
  if (n >= N) return;
  const float* xr = x + (size_t)n * K;
  float a0 = 0.f, a1 = 0.f;
  for (int k = 0; k < K; ++k) {
    float xv = xr[k];
    a0 += xv * wf[k * 2];
    a1 += xv * wf[k * 2 + 1];
  }
  alpha[n * 2] = a0; alpha[n * 2 + 1] = a1;
}

__global__ void fill_f32(float* __restrict__ p, float v, int n) {
  int i = blockIdx.x * blockDim.x + threadIdx.x;
  if (i < n) p[i] = v;
}

__device__ inline void atomicMaxF(float* addr, float val) {
  if (val >= 0.0f) atomicMax((int*)addr, __float_as_int(val));
  else             atomicMin((unsigned int*)addr, __float_as_uint(val));
}

// Pass 1: leaky-relu logits per edge + running segment max over dst
__global__ void edge_logits_max(const int* __restrict__ src, const int* __restrict__ dst,
                                const float* __restrict__ as_, const float* __restrict__ ad_,
                                float* __restrict__ elog, float* __restrict__ mmax, int E) {
  int e = blockIdx.x * blockDim.x + threadIdx.x;
  if (e >= E) return;
  int s = src[e], d = dst[e];
#pragma unroll
  for (int h = 0; h < 2; ++h) {
    float v = as_[s * 2 + h] + ad_[d * 2 + h];
    v = (v > 0.0f) ? v : NEG_SLOPE * v;
    elog[e * 2 + h] = v;
    atomicMaxF(&mmax[d * 2 + h], v);
  }
}

// Pass 2: p = exp(e - max[dst]); segment sum over dst
__global__ void edge_softmax_p(const int* __restrict__ dst, float* __restrict__ elog,
                               const float* __restrict__ mmax, float* __restrict__ ssum,
                               int E) {
  int e = blockIdx.x * blockDim.x + threadIdx.x;
  if (e >= E) return;
  int d = dst[e];
#pragma unroll
  for (int h = 0; h < 2; ++h) {
    float p = expf(elog[e * 2 + h] - mmax[d * 2 + h]);
    elog[e * 2 + h] = p;
    atomicAdd(&ssum[d * 2 + h], p);
  }
}

// Pass 3: out[dst] += (p/sum) * hs[src]   (one wave per edge, 256 cols)
__global__ void edge_aggregate(const int* __restrict__ src, const int* __restrict__ dst,
                               const float* __restrict__ p, const float* __restrict__ ssum,
                               const _Float16* __restrict__ hs, float* __restrict__ outacc,
                               int E) {
  int g = blockIdx.x * blockDim.x + threadIdx.x;
  int e = g >> 5, lane = g & 31;
  if (e >= E) return;
  int s = src[e], d = dst[e];
  float w0 = p[e * 2]     / (ssum[d * 2]     + 1e-16f);
  float w1 = p[e * 2 + 1] / (ssum[d * 2 + 1] + 1e-16f);
  const _Float16* hr = hs + (size_t)s * 256;
  float* orow = outacc + (size_t)d * 256;
#pragma unroll
  for (int it = 0; it < 8; ++it) {
    int idx = it * 32 + lane;
    float v = (float)hr[idx] * ((idx < 128) ? w0 : w1);
    atomicAdd(&orow[idx], v);
  }
}

// x = relu(acc + b1 [+ b2])
__global__ void combine_relu(const float* __restrict__ acc, const float* __restrict__ b1,
                             const float* __restrict__ b2, float* __restrict__ xo, int N) {
  int i = blockIdx.x * blockDim.x + threadIdx.x;
  if (i >= N * 256) return;
  int c = i & 255;
  float v = acc[i] + b1[c] + (b2 ? b2[c] : 0.0f);
  xo[i] = v > 0.0f ? v : 0.0f;
}

// Final per-edge MLP: sigmoid([x_src | ea | x_dst] @ lin_w + lin_b), wave/edge
__global__ void edge_mlp(const int* __restrict__ src, const int* __restrict__ dst,
                         const float* __restrict__ xsrc, const float* __restrict__ xdst,
                         const float* __restrict__ ea, const float* __restrict__ lin_w,
                         const float* __restrict__ lin_b, float* __restrict__ out,
                         int E, int obase) {
  int g = blockIdx.x * blockDim.x + threadIdx.x;
  int e = g >> 5, lane = g & 31;
  if (e >= E) return;
  int s = src[e], d = dst[e];
  float a0 = 0.f, a1 = 0.f;
  for (int j = lane; j < 520; j += 32) {
    float fv;
    if (j < 256)      fv = xsrc[(size_t)s * 256 + j];
    else if (j < 264) fv = ea[(size_t)e * 8 + (j - 256)];
    else              fv = xdst[(size_t)d * 256 + (j - 264)];
    a0 += fv * lin_w[j * 2];
    a1 += fv * lin_w[j * 2 + 1];
  }
#pragma unroll
  for (int m = 16; m >= 1; m >>= 1) {
    a0 += __shfl_xor(a0, m, 32);
    a1 += __shfl_xor(a1, m, 32);
  }
  if (lane == 0) {
    out[(size_t)(obase + e) * 2]     = 1.0f / (1.0f + expf(-(a0 + lin_b[0])));
    out[(size_t)(obase + e) * 2 + 1] = 1.0f / (1.0f + expf(-(a1 + lin_b[1])));
  }
}

// ---------------------------------------------------------------------------
// Host orchestration
// ---------------------------------------------------------------------------
static inline int cdiv(int a, int b) { return (a + b - 1) / b; }

struct ConvBufs {
  float *foldS, *foldD, *AS, *AD, *elog, *m, *s;
};

static void run_conv(hipStream_t st,
                     const float* xs, int Ns, int Ks,
                     const float* xd, int Nd, int Kd,
                     const float* w_s, const float* w_d,
                     const float* att_s, const float* att_d,
                     const int* src, const int* dst, int E,
                     _Float16* hs, float* outacc, const ConvBufs& B) {
  fold_att<<<cdiv(Ks, 256), 256, 0, st>>>(w_s, att_s, B.foldS, Ks);
  fold_att<<<cdiv(Kd, 256), 256, 0, st>>>(w_d, att_d, B.foldD, Kd);
  node_alpha<<<cdiv(Ns, 256), 256, 0, st>>>(xs, B.foldS, B.AS, Ns, Ks);
  node_alpha<<<cdiv(Nd, 256), 256, 0, st>>>(xd, B.foldD, B.AD, Nd, Kd);
  wmma_gemm_f16<<<cdiv(Ns, 32), 256, 0, st>>>(xs, w_s, hs, Ns, Ks);
  fill_f32<<<cdiv(Nd * 2, 256), 256, 0, st>>>(B.m, -INFINITY, Nd * 2);
  fill_f32<<<cdiv(Nd * 2, 256), 256, 0, st>>>(B.s, 0.0f, Nd * 2);
  edge_logits_max<<<cdiv(E, 256), 256, 0, st>>>(src, dst, B.AS, B.AD, B.elog, B.m, E);
  edge_softmax_p<<<cdiv(E, 256), 256, 0, st>>>(dst, B.elog, B.m, B.s, E);
  edge_aggregate<<<cdiv(E * 32, 256), 256, 0, st>>>(src, dst, B.elog, B.s, hs, outacc, E);
}

extern "C" void kernel_launch(void* const* d_in, const int* in_sizes, int n_in,
                              void* d_out, int out_size, void* d_ws, size_t ws_size,
                              hipStream_t stream) {
  (void)in_sizes; (void)n_in; (void)out_size; (void)ws_size;
  const float* x_req = (const float*)d_in[0];
  const float* x_veh = (const float*)d_in[1];
  const float* ea_rr = (const float*)d_in[2];
  const float* ea_vr = (const float*)d_in[3];
  const float* ea_rv = (const float*)d_in[4];
  const float *ws_[6], *wd_[6], *as_[6], *ad_[6], *b_[6];
  for (int t = 0; t < 6; ++t) {
    ws_[t] = (const float*)d_in[5 + t * 5 + 0];
    wd_[t] = (const float*)d_in[5 + t * 5 + 1];
    as_[t] = (const float*)d_in[5 + t * 5 + 2];
    ad_[t] = (const float*)d_in[5 + t * 5 + 3];
    b_[t]  = (const float*)d_in[5 + t * 5 + 4];
  }
  const float* lin_w = (const float*)d_in[35];
  const float* lin_b = (const float*)d_in[36];
  const int* e_rr = (const int*)d_in[37];
  const int* e_vr = (const int*)d_in[38];
  const int* e_rv = (const int*)d_in[39];
  const int *src_rr = e_rr, *dst_rr = e_rr + E_RR;
  const int *src_vr = e_vr, *dst_vr = e_vr + E_VR;
  const int *src_rv = e_rv, *dst_rv = e_rv + E_RV;
  float* out = (float*)d_out;

  // carve workspace
  char* base = (char*)d_ws;
  size_t off = 0;
  auto carve = [&](size_t bytes) -> void* {
    off = (off + 255) & ~(size_t)255;
    void* r = base + off;
    off += bytes;
    return r;
  };
  _Float16* hs_big  = (_Float16*)carve((size_t)N_REQ * 256 * 2);
  _Float16* hs_sml  = (_Float16*)carve((size_t)N_VEH * 256 * 2);
  float* out_req = (float*)carve((size_t)N_REQ * 256 * 4);
  float* out_veh = (float*)carve((size_t)N_VEH * 256 * 4);
  float* xr1 = (float*)carve((size_t)N_REQ * 256 * 4);
  float* xv1 = (float*)carve((size_t)N_VEH * 256 * 4);
  float* xr2 = (float*)carve((size_t)N_REQ * 256 * 4);
  float* xv2 = (float*)carve((size_t)N_VEH * 256 * 4);
  ConvBufs B;
  B.elog  = (float*)carve((size_t)E_RR * 2 * 4);
  B.AS    = (float*)carve((size_t)N_REQ * 2 * 4);
  B.AD    = (float*)carve((size_t)N_REQ * 2 * 4);
  B.m     = (float*)carve((size_t)N_REQ * 2 * 4);
  B.s     = (float*)carve((size_t)N_REQ * 2 * 4);
  B.foldS = (float*)carve(256 * 2 * 4);
  B.foldD = (float*)carve(256 * 2 * 4);

  hipStream_t st = stream;
  const int NR = N_REQ * 256, NV = N_VEH * 256;

  // ---- layer 1 ----
  fill_f32<<<cdiv(NR, 256), 256, 0, st>>>(out_req, 0.0f, NR);
  fill_f32<<<cdiv(NV, 256), 256, 0, st>>>(out_veh, 0.0f, NV);
  run_conv(st, x_req, N_REQ, 64, x_req, N_REQ, 64, ws_[0], wd_[0], as_[0], ad_[0],
           src_rr, dst_rr, E_RR, hs_big, out_req, B);
  run_conv(st, x_veh, N_VEH, 32, x_req, N_REQ, 64, ws_[1], wd_[1], as_[1], ad_[1],
           src_vr, dst_vr, E_VR, hs_sml, out_req, B);
  run_conv(st, x_req, N_REQ, 64, x_veh, N_VEH, 32, ws_[2], wd_[2], as_[2], ad_[2],
           src_rv, dst_rv, E_RV, hs_big, out_veh, B);
  combine_relu<<<cdiv(NR, 256), 256, 0, st>>>(out_req, b_[0], b_[1], xr1, N_REQ);
  combine_relu<<<cdiv(NV, 256), 256, 0, st>>>(out_veh, b_[2], nullptr, xv1, N_VEH);

  // ---- layer 2 ----
  fill_f32<<<cdiv(NR, 256), 256, 0, st>>>(out_req, 0.0f, NR);
  fill_f32<<<cdiv(NV, 256), 256, 0, st>>>(out_veh, 0.0f, NV);
  run_conv(st, xr1, N_REQ, 256, xr1, N_REQ, 256, ws_[3], wd_[3], as_[3], ad_[3],
           src_rr, dst_rr, E_RR, hs_big, out_req, B);
  run_conv(st, xv1, N_VEH, 256, xr1, N_REQ, 256, ws_[4], wd_[4], as_[4], ad_[4],
           src_vr, dst_vr, E_VR, hs_sml, out_req, B);
  run_conv(st, xr1, N_REQ, 256, xv1, N_VEH, 256, ws_[5], wd_[5], as_[5], ad_[5],
           src_rv, dst_rv, E_RV, hs_big, out_veh, B);
  combine_relu<<<cdiv(NR, 256), 256, 0, st>>>(out_req, b_[3], b_[4], xr2, N_REQ);
  combine_relu<<<cdiv(NV, 256), 256, 0, st>>>(out_veh, b_[5], nullptr, xv2, N_VEH);

  // ---- final per-edge MLP + sigmoid ----
  edge_mlp<<<cdiv(E_RR * 32, 256), 256, 0, st>>>(src_rr, dst_rr, xr2, xr2, ea_rr,
                                                 lin_w, lin_b, out, E_RR, 0);
  edge_mlp<<<cdiv(E_VR * 32, 256), 256, 0, st>>>(src_vr, dst_vr, xv2, xr2, ea_vr,
                                                 lin_w, lin_b, out, E_VR, E_RR);
  edge_mlp<<<cdiv(E_RV * 32, 256), 256, 0, st>>>(src_rv, dst_rv, xr2, xv2, ea_rv,
                                                 lin_w, lin_b, out, E_RV, E_RR + E_VR);
}
